// RegionProposalNetwork_74612171866712
// MI455X (gfx1250) — matrix operand order repaired and checked by reading, not compiled, not run on hardware
//
#include <hip/hip_runtime.h>

typedef _Float16 v8h  __attribute__((ext_vector_type(8)));
typedef _Float16 v16h __attribute__((ext_vector_type(16)));
typedef float    v8f  __attribute__((ext_vector_type(8)));
typedef unsigned int v4u __attribute__((ext_vector_type(4)));
typedef int      v8i  __attribute__((ext_vector_type(8)));
typedef int      v4i  __attribute__((ext_vector_type(4)));

union Frag { v16h f; v8h h[2]; };

#define NPOS     2500     // 50*50
#define NANC     22500    // 2500*9
#define SORTN    32768
#define PRE_NMS_N 6000
#define POST_NMS_N 300
#define PADW     52       // 50 + 2 halo
#define PADS     2704     // 52*52
#define IC       2048
#define OC       1024
#define KCHUNK   128      // K-halfs staged per TDM chunk
#define CHUNKS_PER_R (IC / KCHUNK)     // 16
#define NCHUNKS  (9 * CHUNKS_PER_R)    // 144

__device__ __forceinline__ float ninf() { return -__builtin_huge_valf(); }

// ---------------- pre-pass: weights fp32 (1024,2048,3,3) -> f16 [tap][oc][ic]
__global__ void cvt_weights(const float* __restrict__ W1, _Float16* __restrict__ Wt, int n) {
    int i = blockIdx.x * blockDim.x + threadIdx.x;
    if (i >= n) return;
    int r   = i / (OC * IC);
    int rem = i - r * (OC * IC);
    int oc  = rem / IC;
    int ic  = rem - oc * IC;
    Wt[i] = (_Float16)W1[(size_t)(oc * IC + ic) * 9 + r];
}

// ---------------- pre-pass: features fp32 (2048,50,50) -> f16 [padded spatial 52x52][ic]
__global__ void cvt_features(const float* __restrict__ F, _Float16* __restrict__ Ft, int n) {
    int i = blockIdx.x * blockDim.x + threadIdx.x;
    if (i >= n) return;
    int s  = i / IC;
    int ic = i - s * IC;
    int yy = s / PADW - 1;
    int xx = s % PADW - 1;
    float v = (yy >= 0 && yy < 50 && xx >= 0 && xx < 50)
                  ? F[((size_t)ic * 50 + yy) * 50 + xx] : 0.0f;
    Ft[i] = (_Float16)v;
}

// ---------------- 3x3 conv as implicit GEMM:
//   A (weights) staged in LDS by TDM tensor_load_to_lds, double-buffered across chunks,
//   A and B fragments register ping-ponged, prefetch groups pinned by sched_barrier.
// grid (16, 40), block 128 (4 waves). Wave = 64(M) x 16(N) strip.
__global__ void __launch_bounds__(128, 1)
conv3x3_wmma(const _Float16* __restrict__ Wt,
             const _Float16* __restrict__ Ft,
             const float* __restrict__ b1,
             float* __restrict__ x_t) {
    __shared__ _Float16 shA[2][64 * KCHUNK];       // 2 x 16 KB

    const int lane  = threadIdx.x & 31;
    const int wave  = threadIdx.x >> 5;
    const int mbase = blockIdx.x * 64;
    const int ntile = blockIdx.y * 4 + wave;
    const bool live = (ntile < 157);               // wave-uniform

    const int nl    = lane & 15;                   // A row m%16 / B col n%16 / C col
    const int khalf = lane >> 4;
    const int n     = ntile * 16 + nl;

    int s = PADW + 1;                              // safe interior dummy
    if (live && n < NPOS) { int y = n / 50, xx = n - y * 50; s = (y + 1) * PADW + (xx + 1); }

    // B fragment lane base: K = kbase + khalf*8 .. +7 and kbase + 16 + khalf*8 .. +7 (ISA B 32x16 16-bit layout)
    const _Float16* bbase = Ft + (size_t)s * IC + khalf * 8;

    const unsigned ldsbase = (unsigned)(size_t)(void*)&shA[0][0];   // low 32 bits = LDS byte addr

    v8f acc[4];
#pragma unroll
    for (int t = 0; t < 4; ++t) acc[t] = (v8f){0.f,0.f,0.f,0.f,0.f,0.f,0.f,0.f};

    // ---- TDM issue: load 64 rows x 128 halfs of Wt for chunk c into shA[buf]
    auto issue_tdm = [&](int c, int buf) {
        const int r  = c >> 4;
        const int kc = (c & 15) * KCHUNK;
        unsigned long long ga = (unsigned long long)(const void*)Wt
            + ((size_t)((size_t)(r * OC + mbase) * IC + kc) * 2);
        unsigned lds = ldsbase + (unsigned)(buf * 64 * KCHUNK * 2);
        // D# group 0: count=1 | lds_addr | global_addr[56:0] | type=2
        v4u g0;
        g0[0] = 1u;
        g0[1] = lds;
        g0[2] = (unsigned)ga;
        g0[3] = (unsigned)((ga >> 32) & 0x01FFFFFFull) | 0x80000000u;
        // D# group 1: data_size=2B; tensor_dim0=2048, tensor_dim1=big; tile 128x64;
        // tensor_dim0_stride = 2048 elements
        const unsigned DIM1 = 0x100000u;
        v8i g1;
        g1[0] = 0x00010000;                                  // data_size=1 (2 bytes)
        g1[1] = (int)((2048u & 0xFFFFu) << 16);              // tensor_dim0[15:0]
        g1[2] = (int)((2048u >> 16) | ((DIM1 & 0xFFFFu) << 16));
        g1[3] = (int)((DIM1 >> 16) | (128u << 16));          // tile_dim0 = 128
        g1[4] = 64;                                          // tile_dim1 = 64, tile_dim2 = 0
        g1[5] = 2048;                                        // tensor_dim0_stride lo32
        g1[6] = 0;
        g1[7] = 0;
        v4i g2 = {0, 0, 0, 0};
        v4i g3 = {0, 0, 0, 0};
#if __clang_major__ >= 23
        v8i g4 = {0, 0, 0, 0, 0, 0, 0, 0};
        __builtin_amdgcn_tensor_load_to_lds(g0, g1, g2, g3, g4, 0);
#else
        __builtin_amdgcn_tensor_load_to_lds(g0, g1, g2, g3, 0);
#endif
    };

    // prologue: stage chunk 0
    if (wave == 0) {
        issue_tdm(0, 0);
        __builtin_amdgcn_s_wait_tensorcnt((short)0);
    }
    __syncthreads();

    for (int c = 0; c < NCHUNKS; ++c) {
        const int buf = c & 1;
        if (wave == 0 && (c + 1) < NCHUNKS)
            issue_tdm(c + 1, buf ^ 1);                        // overlaps with compute below

        if (live) {
            const int r  = c >> 4;
            const int kc = (c & 15) * KCHUNK;
            const int dy = r / 3 - 1, dx = r % 3 - 1;
            const _Float16* bp = bbase + (dy * PADW + dx) * IC + kc;
            const _Float16* sa = &shA[buf][0] + nl * KCHUNK + khalf * 8;

            __builtin_prefetch(bp + KCHUNK, 0, 1);            // warm next chunk's B

            // A and B register ping-pong: loads run one k-step ahead of the WMMAs.
            Frag abuf[2][4];
            Frag bbuf[2];
            bbuf[0].h[0] = *(const v8h*)(bp);
            bbuf[0].h[1] = *(const v8h*)(bp + 16);
#pragma unroll
            for (int t = 0; t < 4; ++t) {
                const _Float16* at = sa + t * 16 * KCHUNK;
                abuf[0][t].h[0] = *(const v8h*)(at);
                abuf[0][t].h[1] = *(const v8h*)(at + 16);
            }
            __builtin_amdgcn_sched_barrier(0);
#pragma unroll
            for (int q = 0; q < 4; ++q) {
                if (q < 3) {
                    const int kb2 = (q + 1) * 32;
                    bbuf[(q + 1) & 1].h[0] = *(const v8h*)(bp + kb2);
                    bbuf[(q + 1) & 1].h[1] = *(const v8h*)(bp + kb2 + 16);
#pragma unroll
                    for (int t = 0; t < 4; ++t) {
                        const _Float16* at = sa + t * 16 * KCHUNK + kb2;
                        abuf[(q + 1) & 1][t].h[0] = *(const v8h*)(at);
                        abuf[(q + 1) & 1][t].h[1] = *(const v8h*)(at + 16);
                    }
                }
                __builtin_amdgcn_sched_barrier(0);            // loads(q+1) stay above wmmas(q)
#pragma unroll
                for (int t = 0; t < 4; ++t)
                    acc[t] = __builtin_amdgcn_wmma_f32_16x16x32_f16(
                        false, abuf[q & 1][t].f, false, bbuf[q & 1].f, (short)0, acc[t], false, false);
                __builtin_amdgcn_sched_barrier(0);            // wmmas(q) stay above loads(q+2)
            }
        }

        if (wave == 0) __builtin_amdgcn_s_wait_tensorcnt((short)0);
        __syncthreads();
    }

    if (live && n < NPOS) {
#pragma unroll
        for (int t = 0; t < 4; ++t) {
#pragma unroll
            for (int g = 0; g < 8; ++g) {
                int m = mbase + t * 16 + g + khalf * 8;
                x_t[(size_t)n * OC + m] = acc[t][g] + b1[m];
            }
        }
    }
}

// ---------------- 1x1 heads: per pixel, 54 dot products over 1024 channels
__global__ void heads_kernel(const float* __restrict__ x_t,
                             const float* __restrict__ Wreg, const float* __restrict__ breg,
                             const float* __restrict__ Wcls, const float* __restrict__ bcls,
                             float* __restrict__ head) {
    __shared__ float xv[OC];
    const int p = blockIdx.x;
    const int tid = threadIdx.x;
    for (int c = tid; c < OC; c += blockDim.x) xv[c] = x_t[(size_t)p * OC + c];
    __syncthreads();
    const int lane = tid & 31, wv = tid >> 5;
    for (int ch = wv; ch < 54; ch += 8) {
        const float* wrow = (ch < 36) ? (Wreg + (size_t)ch * OC) : (Wcls + (size_t)(ch - 36) * OC);
        float sum = 0.f;
        for (int c = lane; c < OC; c += 32) sum += wrow[c] * xv[c];
        for (int off = 16; off > 0; off >>= 1) sum += __shfl_down(sum, off, 32);
        if (lane == 0)
            head[(size_t)p * 54 + ch] = sum + ((ch < 36) ? breg[ch] : bcls[ch - 36]);
    }
}

// ---------------- decode + min-size filter + sort-key init (padded to SORTN)
__global__ void decode_kernel(const float* __restrict__ head,
                              const float* __restrict__ anchors,
                              float* __restrict__ boxes,
                              float* __restrict__ keys, int* __restrict__ idxs) {
    int i = blockIdx.x * blockDim.x + threadIdx.x;
    if (i >= SORTN) return;
    if (i < NANC) {
        int p = i / 9, a = i - p * 9;
        const float* l = head + (size_t)p * 54 + a * 4;
        float score = head[(size_t)p * 54 + 36 + a * 2 + 1];
        float a0 = anchors[i * 4 + 0], a1 = anchors[i * 4 + 1];
        float a2 = anchors[i * 4 + 2], a3 = anchors[i * 4 + 3];
        float ah = a2 - a0, aw = a3 - a1;
        float acy = a0 + 0.5f * ah, acx = a1 + 0.5f * aw;
        float cy = l[0] * ah + acy, cx = l[1] * aw + acx;
        float h = __expf(l[2]) * ah, w = __expf(l[3]) * aw;
        float y1 = fminf(fmaxf(cy - 0.5f * h, 0.f), 800.f);
        float x1 = fminf(fmaxf(cx - 0.5f * w, 0.f), 800.f);
        float y2 = fminf(fmaxf(cy + 0.5f * h, 0.f), 800.f);
        float x2 = fminf(fmaxf(cx + 0.5f * w, 0.f), 800.f);
        if (!((y2 - y1 >= 16.f) && (x2 - x1 >= 16.f))) score = ninf();
        boxes[i * 4 + 0] = y1; boxes[i * 4 + 1] = x1;
        boxes[i * 4 + 2] = y2; boxes[i * 4 + 3] = x2;
        keys[i] = score; idxs[i] = i;
    } else {
        keys[i] = ninf(); idxs[i] = 0x7fffffff;
    }
}

// ---------------- bitonic sort pass (descending; idx tiebreak = stable argsort(-score))
__global__ void bitonic_pass(float* __restrict__ keys, int* __restrict__ idxs, int j, int k) {
    int i = blockIdx.x * blockDim.x + threadIdx.x;
    int ixj = i ^ j;
    if (ixj > i) {
        float ki = keys[i], kj = keys[ixj];
        int ii = idxs[i], ij = idxs[ixj];
        bool firstWins = (ki > kj) || (ki == kj && ii < ij);
        bool desc = ((i & k) == 0);
        if (desc ? !firstWins : firstWins) {
            keys[i] = kj; keys[ixj] = ki;
            idxs[i] = ij; idxs[ixj] = ii;
        }
    }
}

// ---------------- gather top-6000 boxes + validity
__global__ void gather_kernel(const float* __restrict__ boxes,
                              const float* __restrict__ keys, const int* __restrict__ idxs,
                              float* __restrict__ sb, int* __restrict__ valid0) {
    int j = blockIdx.x * blockDim.x + threadIdx.x;
    if (j >= PRE_NMS_N) return;
    int src = idxs[j];
    sb[j * 4 + 0] = boxes[src * 4 + 0];
    sb[j * 4 + 1] = boxes[src * 4 + 1];
    sb[j * 4 + 2] = boxes[src * 4 + 2];
    sb[j * 4 + 3] = boxes[src * 4 + 3];
    valid0[j] = (keys[j] > ninf()) ? 1 : 0;
}

// ---------------- sequential NMS, one block of 1024 threads
__global__ void nms_kernel(const float* __restrict__ sb, const int* __restrict__ valid0,
                           int* __restrict__ keep) {
    __shared__ unsigned char valid[PRE_NMS_N];
    __shared__ int redidx[32];
    __shared__ int chosen;
    const int tid = threadIdx.x, lane = tid & 31, wv = tid >> 5;
    for (int j = tid; j < PRE_NMS_N; j += 1024) valid[j] = (unsigned char)valid0[j];
    __syncthreads();
    for (int it = 0; it < POST_NMS_N; ++it) {
        int mi = 0x7fffffff;
        for (int j = tid; j < PRE_NMS_N; j += 1024)
            if (valid[j]) { mi = j; break; }
        for (int off = 16; off > 0; off >>= 1) {
            int o = __shfl_down(mi, off, 32);
            mi = (o < mi) ? o : mi;
        }
        if (lane == 0) redidx[wv] = mi;
        __syncthreads();
        if (tid == 0) {
            int c = 0x7fffffff;
            for (int w = 0; w < 32; ++w) c = (redidx[w] < c) ? redidx[w] : c;
            chosen = c;
            keep[it] = (c == 0x7fffffff) ? -1 : c;
        }
        __syncthreads();
        int c = chosen;
        if (c != 0x7fffffff) {
            float by1 = sb[c * 4 + 0], bx1 = sb[c * 4 + 1];
            float by2 = sb[c * 4 + 2], bx2 = sb[c * 4 + 3];
            float barea = (bx2 - bx1 + 1.f) * (by2 - by1 + 1.f);
            for (int j = tid; j < PRE_NMS_N; j += 1024) {
                if (!valid[j]) continue;
                float y1 = sb[j * 4 + 0], x1 = sb[j * 4 + 1];
                float y2 = sb[j * 4 + 2], x2 = sb[j * 4 + 3];
                float area = (x2 - x1 + 1.f) * (y2 - y1 + 1.f);
                float yy1 = fmaxf(by1, y1), xx1 = fmaxf(bx1, x1);
                float yy2 = fminf(by2, y2), xx2 = fminf(bx2, x2);
                float inter = fmaxf(0.f, xx2 - xx1 + 1.f) * fmaxf(0.f, yy2 - yy1 + 1.f);
                float iou = inter / (barea + area - inter);
                if (iou > 0.7f) valid[j] = 0;
            }
        }
        __syncthreads();
    }
}

// ---------------- emit rois (300,4)
__global__ void write_rois(const float* __restrict__ sb, const int* __restrict__ keep,
                           float* __restrict__ out) {
    int i = blockIdx.x * blockDim.x + threadIdx.x;
    if (i >= POST_NMS_N * 4) return;
    int r = i >> 2, d = i & 3;
    int k = keep[r];
    out[i] = (k >= 0) ? sb[k * 4 + d] : 0.0f;
}

extern "C" void kernel_launch(void* const* d_in, const int* in_sizes, int n_in,
                              void* d_out, int out_size, void* d_ws, size_t ws_size,
                              hipStream_t stream) {
    const float* features = (const float*)d_in[0];
    const float* anchors  = (const float*)d_in[1];
    const float* W1       = (const float*)d_in[2];
    const float* b1       = (const float*)d_in[3];
    const float* Wreg     = (const float*)d_in[4];
    const float* breg     = (const float*)d_in[5];
    const float* Wcls     = (const float*)d_in[6];
    const float* bcls     = (const float*)d_in[7];

    char* ws = (char*)d_ws;
    size_t off = 0;
    auto carve = [&](size_t bytes) -> void* {
        void* p = ws + off;
        off = (off + bytes + 255) & ~(size_t)255;
        return p;
    };
    _Float16* Wt    = (_Float16*)carve((size_t)9 * OC * IC * 2);   // 37.75 MB
    _Float16* Ft    = (_Float16*)carve((size_t)PADS * IC * 2);     // 11.08 MB
    float*    x_t   = (float*)carve((size_t)NPOS * OC * 4);        // 10.24 MB
    float*    head  = (float*)carve((size_t)NPOS * 54 * 4);
    float*    boxes = (float*)carve((size_t)NANC * 4 * 4);
    float*    keys  = (float*)carve((size_t)SORTN * 4);
    int*      idxs  = (int*)carve((size_t)SORTN * 4);
    float*    sb    = (float*)carve((size_t)PRE_NMS_N * 4 * 4);
    int*      valid0= (int*)carve((size_t)PRE_NMS_N * 4);
    int*      keep  = (int*)carve((size_t)POST_NMS_N * 4);

    {   // f32 -> f16 conversions
        int nw = 9 * OC * IC;
        cvt_weights<<<(nw + 255) / 256, 256, 0, stream>>>(W1, Wt, nw);
        int nf = PADS * IC;
        cvt_features<<<(nf + 255) / 256, 256, 0, stream>>>(features, Ft, nf);
    }

    // 3x3 conv: 16 M-tiles of 64 x 157 N-tiles of 16 (4 waves/block), TDM-staged A
    conv3x3_wmma<<<dim3(16, 40), 128, 0, stream>>>(Wt, Ft, b1, x_t);

    heads_kernel<<<NPOS, 256, 0, stream>>>(x_t, Wreg, breg, Wcls, bcls, head);

    decode_kernel<<<SORTN / 256, 256, 0, stream>>>(head, anchors, boxes, keys, idxs);

    for (int k = 2; k <= SORTN; k <<= 1)
        for (int j = k >> 1; j >= 1; j >>= 1)
            bitonic_pass<<<SORTN / 256, 256, 0, stream>>>(keys, idxs, j, k);

    gather_kernel<<<(PRE_NMS_N + 255) / 256, 256, 0, stream>>>(boxes, keys, idxs, sb, valid0);

    nms_kernel<<<1, 1024, 0, stream>>>(sb, valid0, keep);

    write_rois<<<(POST_NMS_N * 4 + 255) / 256, 256, 0, stream>>>(sb, keep, (float*)d_out);
}
